// NeRFBackbone_91122026152338
// MI455X (gfx1250) — matrix operand order repaired and checked by reading, not compiled, not run on hardware
//
#include <hip/hip_runtime.h>
#include <hip/hip_bf16.h>
#include <stdint.h>

// Flip to 0 if the gfx1250 assembler rejects the async-to-LDS mnemonics.
#define ASYNC_STAGE 1

typedef _Float16 v16h __attribute__((ext_vector_type(16)));
typedef _Float16 v8h  __attribute__((ext_vector_type(8)));
typedef float    v8f  __attribute__((ext_vector_type(8)));

union AFrag { v16h v; v8h h[2]; };

// ---------------- layer tables (11 WMMA layers: dW0..dW7, cW0..cW2) ----------
// frag counts: 24,32,32,32,32,56,32,32,20,8,8  (total 308)
constexpr int NLAYER = 11;
constexpr int FRAG_BASE[NLAYER] = {0,24,56,88,120,152,208,240,272,292,300};
constexpr int KTILES[NLAYER]    = {3,4,4,4,4,7,4,4,5,2,2};   // K = 32*KT (padded)
constexpr int NTILES[NLAYER]    = {8,8,8,8,8,8,8,8,4,4,4};   // N = 16*NT
// padded-k -> source-row mapping: seg0 [0,len0) -> row k ; seg1 [off1,off1+len1) -> sb1+(k-off1)
constexpr int SEG0[NLAYER]      = {67,128,128,128,128, 67,128,128,131,64,64};
constexpr int SEG1OFF[NLAYER]   = { 0,  0,  0,  0,  0, 96,  0,  0,  0, 0, 0};
constexpr int SEG1LEN[NLAYER]   = { 0,  0,  0,  0,  0,128,  0,  0,  0, 0, 0};
constexpr int SEG1SRC[NLAYER]   = { 0,  0,  0,  0,  0, 67,  0,  0,  0, 0, 0};
// activation buffer: 16 rows x 256 cols f16 per tile.
// cols [0,96)=x0(pad), [96,224)=h, [224,227)=view, rest zero.
constexpr int ABASE[NLAYER]     = { 0,96,96,96,96, 0,96,96,96, 0, 0};
constexpr int OBASE[NLAYER]     = {96,96,96,96,96,96,96,96, 0, 0, 0};

constexpr int WBUF_HALVES = 56 * 512;          // largest layer (dW5) = 56 frags
constexpr int TILE_HALVES = 16 * 256;          // one tile's activation rows
constexpr int TILES_PER_WAVE = 2;
constexpr int WAVES = 8;
constexpr int TILES_PER_BLOCK = WAVES * TILES_PER_WAVE;   // 16 -> 256 points/block

struct KParams {
  const float* pos; const float* cond; const float* view;
  const float* W[NLAYER]; const float* B[NLAYER];
  const float* sW; const float* sb; const float* rW; const float* rb;
  _Float16* ws; float* out;
};

// ---------------- prepass: f32 weights -> f16 B-fragments in d_ws ------------
__global__ __launch_bounds__(256) void nerf_prep(KParams p) {
  const int f = blockIdx.x;                    // fragment index 0..307
  int L = 0;
  while (L + 1 < NLAYER && f >= FRAG_BASE[L + 1]) ++L;
  const int local = f - FRAG_BASE[L];
  const int KT = KTILES[L], NT = NTILES[L];
  const int nt = local / KT, kt = local % KT;  // frag = nt*KT + kt
  const int Nout = NT * 16;
  const float* W = p.W[L];
  const int len0 = SEG0[L], off1 = SEG1OFF[L], len1 = SEG1LEN[L], sb1 = SEG1SRC[L];
  for (int e = threadIdx.x; e < 512; e += 256) {
    const int lane = e >> 4, h = e & 15;
    const int n = nt * 16 + (lane & 15);               // B col = lane&15
    const int k = kt * 32 + ((lane >> 4) << 3) + (h & 7) + ((h >> 3) << 4);    // B row: lane<16 = K{0..7,16..23}; lane>=16 = K{8..15,24..31}
    int row = -1;
    if (k < len0) row = k;
    else if (len1 && k >= off1 && k < off1 + len1) row = sb1 + (k - off1);
    const float v = (row >= 0) ? W[(size_t)row * Nout + n] : 0.0f;
    p.ws[(size_t)f * 512 + e] = (_Float16)v;
  }
}

// ---------------- one WMMA layer for one 16-point tile -----------------------
template <int KT, int NT>
__device__ __forceinline__ void layer_wmma(_Float16* reg, const _Float16* wbuf,
                                           int aBase, int oBase,
                                           const float* bias, int lane) {
  const int n  = lane & 15;     // A row index == C/D column index
  const int hi = lane >> 4;
  AFrag a[KT];
#pragma unroll
  for (int t = 0; t < KT; ++t) {
    // A 16x32 f16: lane<16 halves = K{0..7,16..23}; lane>=16 = K{8..15,24..31}
    const _Float16* rp = reg + n * 256 + aBase + t * 32 + hi * 8;
    a[t].h[0] = *(const v8h*)rp;
    a[t].h[1] = *(const v8h*)(rp + 16);
  }
#pragma unroll
  for (int nt = 0; nt < NT; ++nt) {
    v8f c = {};
#pragma unroll
    for (int t = 0; t < KT; ++t) {
      AFrag b;
      const _Float16* fp = wbuf + (nt * KT + t) * 512 + lane * 16;
      b.h[0] = *(const v8h*)fp;
      b.h[1] = *(const v8h*)(fp + 8);
      c = __builtin_amdgcn_wmma_f32_16x16x32_f16(false, a[t].v, false, b.v,
                                                 (short)0, c, false, false);
    }
    const float bb = bias[nt * 16 + n];
#pragma unroll
    for (int r = 0; r < 8; ++r) {           // C/D: lane<16 -> M=r, lane>=16 -> M=8+r
      float x = fmaxf(c[r] + bb, 0.0f);     // bias + ReLU fused
      reg[(hi * 8 + r) * 256 + oBase + nt * 16 + n] = (_Float16)x;
    }
  }
}

// ---------------- fused NeRF MLP ---------------------------------------------
__global__ __launch_bounds__(256) void nerf_fused(KParams p) {
  extern __shared__ _Float16 smem[];
  _Float16* wbuf = smem;                       // double-buffered weight frags
  _Float16* abuf = smem + 2 * WBUF_HALVES;     // per-tile activation rows
  const int tid  = threadIdx.x;
  const int lane = tid & 31;
  const int wave = tid >> 5;

  const uint32_t wbufLds = (uint32_t)(uintptr_t)wbuf;  // low 32 bits = LDS offset

  // issue async staging of one layer's fragments into buffer `buf` (no wait)
  auto stage = [&](int L, int buf) {
    const int nChunks = KTILES[L] * NTILES[L] * 64;     // 16B chunks
    const char* src = (const char*)(p.ws + (size_t)FRAG_BASE[L] * 512);
    const uint32_t dstBase = wbufLds + (uint32_t)buf * (WBUF_HALVES * 2);
    for (int j = tid; j < nChunks; j += 256) {
#if ASYNC_STAGE
      uint32_t dst = dstBase + (uint32_t)j * 16u;
      uint64_t ga  = (uint64_t)(uintptr_t)(src + (size_t)j * 16);
      asm volatile("global_load_async_to_lds_b128 %0, %1, off"
                   :: "v"(dst), "v"(ga) : "memory");
#else
      *(uint4*)(wbuf + (size_t)buf * WBUF_HALVES + (size_t)j * 8) =
          *(const uint4*)(src + (size_t)j * 16);
#endif
    }
  };

  // ---- kick off layer-0 weight DMA, then stage inputs while it flies ----
  stage(0, 0);

  for (int t = 0; t < TILES_PER_WAVE; ++t) {
    const int slot = wave * TILES_PER_WAVE + t;
    const int p0   = (blockIdx.x * TILES_PER_BLOCK + slot) * 16;
    _Float16* reg  = abuf + slot * TILE_HALVES;
    uint4 z = make_uint4(0u, 0u, 0u, 0u);
    for (int j = lane; j < TILE_HALVES / 8; j += 32) ((uint4*)reg)[j] = z;
    if (lane < 16) {
#pragma unroll
      for (int c = 0; c < 3; ++c)
        reg[lane * 256 + c] = (_Float16)p.pos[(size_t)(p0 + lane) * 3 + c];
      const int b = p0 >> 7;                   // 128 samples per batch row
#pragma unroll
      for (int c = 0; c < 3; ++c)
        reg[lane * 256 + 224 + c] = (_Float16)p.view[(size_t)b * 3 + c];
    }
    for (int j = lane; j < 16 * 64; j += 32) {
      const int mm = j >> 6, cc = j & 63;
      reg[mm * 256 + 3 + cc] = (_Float16)p.cond[cc];
    }
  }

#if ASYNC_STAGE
  asm volatile("s_wait_asynccnt 0x0" ::: "memory");
#endif
  __syncthreads();                             // layer-0 weights + inputs ready

  for (int L = 0; L < NLAYER; ++L) {
    // prefetch next layer's weights into the other LDS buffer (overlaps WMMA)
    if (L + 1 < NLAYER) stage(L + 1, (L + 1) & 1);

    const _Float16* wb = wbuf + (size_t)(L & 1) * WBUF_HALVES;
    const float* bias = p.B[L];
    for (int t = 0; t < TILES_PER_WAVE; ++t) {
      _Float16* reg = abuf + (wave * TILES_PER_WAVE + t) * TILE_HALVES;
      switch (L) {
        case 0:  layer_wmma<3, 8>(reg, wb, ABASE[0], OBASE[0], bias, lane); break;
        case 5:  layer_wmma<7, 8>(reg, wb, ABASE[5], OBASE[5], bias, lane); break;
        case 8:  layer_wmma<5, 4>(reg, wb, ABASE[8], OBASE[8], bias, lane); break;
        case 9:
        case 10: layer_wmma<2, 4>(reg, wb, 0, 0, bias, lane); break;
        default: layer_wmma<4, 8>(reg, wb, 96, 96, bias, lane); break;
      }
    }

    if (L == 7) {                              // sigma head on h7 (cols 96..223)
      for (int t = 0; t < TILES_PER_WAVE; ++t) {
        const int slot = wave * TILES_PER_WAVE + t;
        const int p0   = (blockIdx.x * TILES_PER_BLOCK + slot) * 16;
        const _Float16* reg = abuf + slot * TILE_HALVES;
        if (lane < 16) {
          float s = p.sb[0];
          const v8h* hp = (const v8h*)(reg + lane * 256 + 96);
#pragma unroll
          for (int c8 = 0; c8 < 16; ++c8) {
            const v8h hv = hp[c8];
#pragma unroll
            for (int e = 0; e < 8; ++e)
              s += (float)hv[e] * p.sW[c8 * 8 + e];
          }
          p.out[(size_t)(p0 + lane) * 4 + 3] = s;
        }
      }
    }

    if (L + 1 < NLAYER) {
#if ASYNC_STAGE
      asm volatile("s_wait_asynccnt 0x0" ::: "memory");
#endif
      __syncthreads();                         // next weights ready, prev reads done
    }
  }

  // ---- rgb head on final color features (cols 0..63) ----
  for (int t = 0; t < TILES_PER_WAVE; ++t) {
    const int slot = wave * TILES_PER_WAVE + t;
    const int p0   = (blockIdx.x * TILES_PER_BLOCK + slot) * 16;
    const _Float16* reg = abuf + slot * TILE_HALVES;
    if (lane < 16) {
      float r0 = p.rb[0], r1 = p.rb[1], r2 = p.rb[2];
      const v8h* hp = (const v8h*)(reg + lane * 256);
#pragma unroll
      for (int c8 = 0; c8 < 8; ++c8) {
        const v8h hv = hp[c8];
#pragma unroll
        for (int e = 0; e < 8; ++e) {
          const float v  = (float)hv[e];
          const int   k  = c8 * 8 + e;
          r0 += v * p.rW[k * 3 + 0];
          r1 += v * p.rW[k * 3 + 1];
          r2 += v * p.rW[k * 3 + 2];
        }
      }
      const size_t o = (size_t)(p0 + lane) * 4;
      p.out[o + 0] = r0; p.out[o + 1] = r1; p.out[o + 2] = r2;
    }
  }
}

// ---------------- host launch ------------------------------------------------
extern "C" void kernel_launch(void* const* d_in, const int* in_sizes, int n_in,
                              void* d_out, int out_size, void* d_ws, size_t ws_size,
                              hipStream_t stream) {
  (void)in_sizes; (void)n_in; (void)out_size; (void)ws_size;
  KParams p;
  p.pos  = (const float*)d_in[0];
  p.cond = (const float*)d_in[1];
  p.view = (const float*)d_in[2];
  const int wIdx[NLAYER] = {3,5,7,9,11,13,15,17,21,23,25};
  const int bIdx[NLAYER] = {4,6,8,10,12,14,16,18,22,24,26};
  for (int i = 0; i < NLAYER; ++i) {
    p.W[i] = (const float*)d_in[wIdx[i]];
    p.B[i] = (const float*)d_in[bIdx[i]];
  }
  p.sW = (const float*)d_in[19]; p.sb = (const float*)d_in[20];
  p.rW = (const float*)d_in[27]; p.rb = (const float*)d_in[28];
  p.ws  = (_Float16*)d_ws;
  p.out = (float*)d_out;

  nerf_prep<<<308, 256, 0, stream>>>(p);       // 308 fragments, one per block

  const size_t shmem = (size_t)(2 * WBUF_HALVES + TILES_PER_BLOCK * TILE_HALVES)
                       * sizeof(_Float16);     // 112KB weights (x2) + 128KB acts
  // 16384 tiles of 16 points, 16 tiles per block
  nerf_fused<<<1024, 256, shmem, stream>>>(p);
}
